// SkillGNN_54700703482252
// MI455X (gfx1250) — compile-verified, matrix operand order checked by reading, not run on hardware
//
#include <hip/hip_runtime.h>
#include <hip/hip_bf16.h>
#include <stdint.h>

// ---------------------------------------------------------------------------
// Types for CDNA5 WMMA (wave32): 16x16x32 bf16 -> f32 accumulate
// ---------------------------------------------------------------------------
typedef __attribute__((ext_vector_type(16))) __bf16 v16bf;
typedef __attribute__((ext_vector_type(8)))  float  v8f;

union BfFrag {
  uint4 u[2];   // 2 x 16B = 16 bf16
  v16bf v;
};

__device__ __forceinline__ unsigned short f2bf(float f) {
  // round-to-nearest-even fp32 -> bf16
  unsigned int u = __float_as_uint(f);
  u += 0x7FFFu + ((u >> 16) & 1u);
  return (unsigned short)(u >> 16);
}

// ---------------------------------------------------------------------------
// Degree / normalization
// ---------------------------------------------------------------------------
__global__ void gcn_deg_init(float* __restrict__ deg, long long n) {
  long long i = (long long)blockIdx.x * blockDim.x + threadIdx.x;
  if (i < n) deg[i] = 1.0f;   // self loop contributes 1
}

__global__ void gcn_deg_count(const long long* __restrict__ cols,
                              float* __restrict__ deg, long long E) {
  long long i = (long long)blockIdx.x * blockDim.x + threadIdx.x;
  if (i < E) atomicAdd(&deg[cols[i]], 1.0f);
}

__global__ void gcn_rsqrt_inplace(float* __restrict__ d, long long n) {
  long long i = (long long)blockIdx.x * blockDim.x + threadIdx.x;
  if (i < n) d[i] = __frsqrt_rn(d[i]);   // deg >= 1 always (self loops)
}

// ---------------------------------------------------------------------------
// fp32 -> bf16 converts (optionally fused ReLU)
// ---------------------------------------------------------------------------
__global__ void gcn_cvt_bf16(const float* __restrict__ in,
                             unsigned short* __restrict__ out, long long n) {
  long long i = (long long)blockIdx.x * blockDim.x + threadIdx.x;
  if (i < n) out[i] = f2bf(in[i]);
}

__global__ void gcn_relu_cvt_bf16(const float* __restrict__ in,
                                  unsigned short* __restrict__ out, long long n) {
  long long i = (long long)blockIdx.x * blockDim.x + threadIdx.x;
  if (i < n) out[i] = f2bf(fmaxf(in[i], 0.0f));
}

// ---------------------------------------------------------------------------
// WMMA GEMM: C[nRows x 128] = A_bf16[nRows x 128] @ W_bf16[128 x 128]^T
//   h[m,n] = sum_k A[m,k] * W[n,k]   (so B-fragment reads W row-major)
// One wave -> one 16x16 tile; block of 8 waves covers 16 rows x 128 cols.
// ---------------------------------------------------------------------------
__global__ __launch_bounds__(256) void gcn_gemm_wmma(
    const unsigned short* __restrict__ A,   // [nRows x 128] bf16 bits
    const unsigned short* __restrict__ W,   // [128 x 128] bf16 bits, row-major
    float* __restrict__ C,                  // [nRows x 128] f32
    int nRows) {
  const int lane  = threadIdx.x & 31;
  const int wave  = threadIdx.x >> 5;       // 0..7 -> output column tile
  const int mBase = blockIdx.x * 16;
  if (mBase >= nRows) return;
  const int nBase = wave * 16;
  const int lrow  = lane & 15;
  const int kb8   = (lane >> 4) * 8;        // K sub-offset per bf16 A/B layout

  int aRow = mBase + lrow;
  if (aRow >= nRows) aRow = nRows - 1;      // tail clamp (stores are guarded)
  const unsigned short* aPtr = A + (size_t)aRow * 128 + kb8;
  const unsigned short* wPtr = W + (size_t)(nBase + lrow) * 128 + kb8;

  v8f c = {};
#pragma unroll
  for (int kb = 0; kb < 128; kb += 32) {
    BfFrag fa, fb;
    fa.u[0] = *(const uint4*)(aPtr + kb);        // K = kb+kb8 .. +7
    fa.u[1] = *(const uint4*)(aPtr + kb + 16);   // K = kb+16+kb8 .. +7
    fb.u[0] = *(const uint4*)(wPtr + kb);
    fb.u[1] = *(const uint4*)(wPtr + kb + 16);
    c = __builtin_amdgcn_wmma_f32_16x16x32_bf16(
        /*neg_a=*/false, fa.v, /*neg_b=*/false, fb.v,
        /*c_mod=*/(short)0, c, /*reuse_a=*/false, /*reuse_b=*/false);
  }

  // C/D layout: VGPR r, lanes 0-15 -> (M=r, N=lane); lanes 16-31 -> (M=r+8)
  const int mOff = (lane >> 4) * 8;
  float* cPtr = C + (size_t)(mBase + mOff) * 128 + nBase + lrow;
  if (mBase + 16 <= nRows) {
    // Fast path (always taken when nRows % 16 == 0): plain stores, no masks.
#pragma unroll
    for (int r = 0; r < 8; ++r) cPtr[(size_t)r * 128] = c[r];
  } else {
#pragma unroll
    for (int r = 0; r < 8; ++r) {
      if (mBase + mOff + r < nRows) cPtr[(size_t)r * 128] = c[r];
    }
  }
}

// ---------------------------------------------------------------------------
// Self-loop + bias (+ optional residual) init:
//   out[n,j] = (residual ? x[n,j] : 0) + bias[j] + inv[n]^2 * h[n,j]
// ---------------------------------------------------------------------------
__global__ void gcn_self_init(const float* __restrict__ h,
                              const float* __restrict__ inv,
                              const float* __restrict__ bias,
                              const float* __restrict__ residual,  // may be null
                              float* __restrict__ out, long long total) {
  long long i = (long long)blockIdx.x * blockDim.x + threadIdx.x;
  if (i >= total) return;
  long long n = i >> 7;
  int j = (int)(i & 127);
  float s = inv[n];
  float v = bias[j] + s * s * h[i];
  if (residual) v += residual[i];
  out[i] = v;
}

// ---------------------------------------------------------------------------
// Edge scatter-add: one wave per edge, 32 lanes x float4 = 128 contiguous
// floats, fully coalesced gather + global_atomic_add_f32 scatter.
// ---------------------------------------------------------------------------
__global__ __launch_bounds__(256) void gcn_edge_agg(
    const long long* __restrict__ rows, const long long* __restrict__ cols,
    const float* __restrict__ h, const float* __restrict__ inv,
    float* __restrict__ out, long long E) {
  long long e = (long long)blockIdx.x * 8 + (threadIdx.x >> 5);
  if (e >= E) return;
  const int lane = threadIdx.x & 31;
  const long long r = rows[e];
  const long long c = cols[e];
  const float norm = inv[r] * inv[c];
  const float4 v = ((const float4*)(h + r * 128))[lane];
  float* o = out + c * 128 + (size_t)lane * 4;
  atomicAdd(o + 0, v.x * norm);
  atomicAdd(o + 1, v.y * norm);
  atomicAdd(o + 2, v.z * norm);
  atomicAdd(o + 3, v.w * norm);
}

// ---------------------------------------------------------------------------
// Host-side orchestration
// ---------------------------------------------------------------------------
extern "C" void kernel_launch(void* const* d_in, const int* in_sizes, int n_in,
                              void* d_out, int out_size, void* d_ws, size_t ws_size,
                              hipStream_t stream) {
  (void)n_in; (void)out_size; (void)ws_size;

  const float*     x  = (const float*)d_in[0];
  const long long* ei = (const long long*)d_in[1];
  const float*     W1 = (const float*)d_in[2];
  const float*     b1 = (const float*)d_in[3];
  const float*     W2 = (const float*)d_in[4];
  const float*     b2 = (const float*)d_in[5];
  float*           out = (float*)d_out;

  const int D = 128;
  const long long N = (long long)in_sizes[0] / D;     // 50000
  const long long E = (long long)in_sizes[1] / 2;     // 1600000
  const long long* rows = ei;        // edge_index[0] : source
  const long long* cols = ei + E;    // edge_index[1] : target

  // Workspace carve-out (256B aligned)
  auto au = [](size_t v) { return (v + 255) & ~(size_t)255; };
  char* ws = (char*)d_ws;
  size_t o = 0;
  float*          inv = (float*)(ws + o);          o += au((size_t)N * 4);
  unsigned short* xb  = (unsigned short*)(ws + o); o += au((size_t)N * D * 2);
  unsigned short* w1b = (unsigned short*)(ws + o); o += au((size_t)D * D * 2);
  unsigned short* w2b = (unsigned short*)(ws + o); o += au((size_t)D * D * 2);
  float*          h   = (float*)(ws + o);          o += au((size_t)N * D * 4);
  float*          agg = (float*)(ws + o);          o += au((size_t)N * D * 4);

  const int TB = 256;
  const long long ND = N * D;
  dim3 bN((unsigned)((N + TB - 1) / TB));
  dim3 bE((unsigned)((E + TB - 1) / TB));
  dim3 bND((unsigned)((ND + TB - 1) / TB));
  dim3 bW((unsigned)((D * D + TB - 1) / TB));
  dim3 bGemm((unsigned)((N + 15) / 16));            // 3125 row tiles
  dim3 bEdge((unsigned)((E + 7) / 8));              // 8 edges (waves) per block

  // 1) symmetric normalization coefficients
  gcn_deg_init<<<bN, TB, 0, stream>>>(inv, N);
  gcn_deg_count<<<bE, TB, 0, stream>>>(cols, inv, E);
  gcn_rsqrt_inplace<<<bN, TB, 0, stream>>>(inv, N);

  // 2) bf16 operands for the WMMA GEMMs
  gcn_cvt_bf16<<<bW, TB, 0, stream>>>(W1, w1b, (long long)D * D);
  gcn_cvt_bf16<<<bW, TB, 0, stream>>>(W2, w2b, (long long)D * D);
  gcn_cvt_bf16<<<bND, TB, 0, stream>>>(x, xb, ND);

  // 3) layer 1: h = x @ W1^T  (v_wmma_f32_16x16x32_bf16)
  gcn_gemm_wmma<<<bGemm, TB, 0, stream>>>(xb, w1b, h, (int)N);

  // 4) agg = b1 + inv^2 * h (self loop), then scatter-add edge messages
  gcn_self_init<<<bND, TB, 0, stream>>>(h, inv, b1, nullptr, agg, ND);
  gcn_edge_agg<<<bEdge, TB, 0, stream>>>(rows, cols, h, inv, agg, E);

  // 5) ReLU + bf16 convert for layer 2 input (reuse xb)
  gcn_relu_cvt_bf16<<<bND, TB, 0, stream>>>(agg, xb, ND);

  // 6) layer 2: h = relu(agg) @ W2^T  (reuse h buffer)
  gcn_gemm_wmma<<<bGemm, TB, 0, stream>>>(xb, w2b, h, (int)N);

  // 7) out = x + b2 + inv^2 * h (residual + self loop), then edge scatter-add
  gcn_self_init<<<bND, TB, 0, stream>>>(h, inv, b2, x, out, ND);
  gcn_edge_agg<<<bEdge, TB, 0, stream>>>(rows, cols, h, inv, out, E);
}
